// Memory_Transformer_54408645705943
// MI455X (gfx1250) — matrix operand order repaired and checked by reading, not compiled, not run on hardware
//
#include <hip/hip_runtime.h>
#include <stdint.h>

// ---------------------------------------------------------------------------
// MI455X (gfx1250) flash-attention decode-prefill step.
// Bandwidth-bound: pre-convert KV to bf16 (halves streamed bytes -> ~6us at
// 23.3 TB/s), WMMA bf16 16x16x32 for QK^T and PV, TDM async tile loads to LDS
// with hardware row padding, double buffered, split-S partials + LSE combine.
// ---------------------------------------------------------------------------

typedef __attribute__((ext_vector_type(16))) __bf16 bf16x16;
typedef __attribute__((ext_vector_type(8)))  __bf16 bf16x8;
typedef __attribute__((ext_vector_type(8)))  float  f32x8;
typedef __attribute__((ext_vector_type(4)))  unsigned int u32x4;
typedef __attribute__((ext_vector_type(8)))  int i32x8;
typedef __attribute__((ext_vector_type(4)))  int i32x4;

constexpr int B   = 1024;
constexpr int S   = 32768;
constexpr int D   = 128;
constexpr int T   = S + B;        // 33792 total keys
constexpr int CH  = 1024;         // keys per S-chunk (grid.y)
constexpr int NCH = T / CH;       // 33 chunks; chunk 32 == causal block
constexpr int BN  = 64;           // keys per inner LDS tile
constexpr int NT  = CH / BN;      // 16 tiles per chunk
constexpr int KROW = D  + 8;      // 136 bf16: 256B row + 16B TDM pad
constexpr int VROW = BN + 8;      // 72  bf16: 128B row + 16B TDM pad
constexpr int PROW = BN + 8;      // P staging row stride
constexpr float SM_SCALE = 0.08838834764831845f; // 1/sqrt(128)

// ---- TDM: 2D tensor tile -> LDS, descriptor per ISA 8.3/8.4 ---------------
__device__ __forceinline__ void tdm_load_2d(
    uint32_t lds_byte_addr, const void* gptr,
    uint32_t td0, uint32_t td1, uint64_t stride0,
    uint32_t tile0, uint32_t tile1,
    uint32_t pad_int_code, uint32_t pad_amt_code)
{
  uint64_t ga = (uint64_t)(uintptr_t)gptr;
  u32x4 g0;
  g0[0] = 1u;                                   // count=1, user descriptor
  g0[1] = lds_byte_addr;                        // lds_addr [63:32]
  g0[2] = (uint32_t)ga;                         // global_addr [95:64]
  g0[3] = (uint32_t)((ga >> 32) & 0x01FFFFFFu)  // global_addr [120:96]
        | (2u << 30);                           // type=2 ("image")
  i32x8 g1;
  g1[0] = (int)((1u << 16)                      // data_size = 2 bytes
              | (1u << 20)                      // pad_enable
              | (pad_int_code << 22)            // pad_interval
              | (pad_amt_code << 25));          // pad_amount
  g1[1] = (int)((td0 & 0xFFFFu) << 16);         // tensor_dim0[15:0]
  g1[2] = (int)((td0 >> 16) | ((td1 & 0xFFFFu) << 16));
  g1[3] = (int)((td1 >> 16) | (tile0 << 16));   // tile_dim0
  g1[4] = (int)(tile1 & 0xFFFFu);               // tile_dim1 (tile_dim2=0)
  g1[5] = (int)(uint32_t)(stride0 & 0xFFFFFFFFu);
  g1[6] = (int)(uint32_t)((stride0 >> 32) & 0xFFFFu);
  g1[7] = 0;
  i32x4 gz = {0, 0, 0, 0};
#if defined(__clang_major__) && (__clang_major__ >= 23)
  i32x8 gz8 = {0, 0, 0, 0, 0, 0, 0, 0};
  __builtin_amdgcn_tensor_load_to_lds(g0, g1, gz, gz, gz8, 0);
#else
  __builtin_amdgcn_tensor_load_to_lds(g0, g1, gz, gz, 0);
#endif
}

__device__ __forceinline__ f32x8 wmma_bf16(bf16x16 a, bf16x16 b, f32x8 c) {
  return __builtin_amdgcn_wmma_f32_16x16x32_bf16(
      /*neg_a=*/false, a, /*neg_b=*/false, b,
      /*c_mod=*/(short)0, c, /*reuse_a=*/false, /*reuse_b=*/false);
}

// ---- Kernel 1: pack K_cache||k -> bf16 row-major --------------------------
__global__ void mtx_pack_k(const float* __restrict__ Kc,
                           const float* __restrict__ knew,
                           __bf16* __restrict__ Kbf) {
  int idx = blockIdx.x * 256 + threadIdx.x;        // < T*D
  int t = idx >> 7;
  float v = (t < S) ? Kc[idx] : knew[idx - S * D];
  Kbf[idx] = (__bf16)v;
}

// ---- Kernel 2: transpose V_cache||v -> Vt bf16 [D][T] ---------------------
__global__ void mtx_transpose_v(const float* __restrict__ Vc,
                                const float* __restrict__ vnew,
                                __bf16* __restrict__ Vt) {
  __shared__ float tile[32][33];
  int t0 = blockIdx.x * 32;
  int d0 = blockIdx.y * 32;
#pragma unroll
  for (int j = 0; j < 4; ++j) {
    int t = t0 + threadIdx.y + j * 8;
    int d = d0 + threadIdx.x;
    float v = (t < S) ? Vc[(size_t)t * D + d] : vnew[(size_t)(t - S) * D + d];
    tile[threadIdx.x][threadIdx.y + j * 8] = v;    // tile[d_local][t_local]
  }
  __syncthreads();
#pragma unroll
  for (int j = 0; j < 4; ++j) {
    int d = d0 + threadIdx.y + j * 8;
    int t = t0 + threadIdx.x;
    Vt[(size_t)d * T + t] = (__bf16)tile[threadIdx.y + j * 8][threadIdx.x];
  }
}

// ---- Kernel 3: flash-attention partials -----------------------------------
// grid.x = 8 Q-tiles of 128 rows (8 waves x 16 rows), grid.y = 33 S-chunks.
__global__ __launch_bounds__(256)
void mtx_attn_partial(const float* __restrict__ q,
                      const __bf16* __restrict__ Kbf,
                      const __bf16* __restrict__ Vt,
                      float* __restrict__ Mb, float* __restrict__ Lb,
                      float* __restrict__ Op) {
  __shared__ __align__(256) __bf16 sK[2][BN * KROW];   // 2 x 17408 B
  __shared__ __align__(256) __bf16 sV[2][D * VROW];    // 2 x 18432 B
  __shared__ __align__(16)  __bf16 sP[8][16 * PROW];   // per-wave P staging

  const int wave = threadIdx.x >> 5;
  const int lane = threadIdx.x & 31;
  const int col  = lane & 15;     // N index / row-within-16
  const int half = lane >> 4;     // hi half of wave
  const int qtile = blockIdx.x;
  const int chunk = blockIdx.y;
  const int qbase = qtile * 128 + wave * 16;
  const int key0  = chunk * CH;
  const bool causal = (key0 + CH) > S;   // only chunk 32

  // LDS byte offsets for TDM (generic LDS pointer low 32 bits = LDS offset)
  const uint32_t aK[2] = { (uint32_t)(uintptr_t)&sK[0][0],
                           (uint32_t)(uintptr_t)&sK[1][0] };
  const uint32_t aV[2] = { (uint32_t)(uintptr_t)&sV[0][0],
                           (uint32_t)(uintptr_t)&sV[1][0] };

  // --- load Q rows into A-operand layout (16x32 bf16 per k-chunk) ---------
  const float* qp = q + (size_t)(qbase + col) * D;
  bf16x16 Qa[4];
#pragma unroll
  for (int c = 0; c < 4; ++c) {
    int base = c * 32 + half * 16;
#pragma unroll
    for (int j = 0; j < 8; ++j) {
      Qa[c][j]     = (__bf16)qp[base + j];
      Qa[c][8 + j] = (__bf16)qp[base + 8 + j];
    }
  }

  float m[8], l[8];
  f32x8 Oacc[8];
#pragma unroll
  for (int r = 0; r < 8; ++r) { m[r] = -1e30f; l[r] = 0.0f; }
#pragma unroll
  for (int d8 = 0; d8 < 8; ++d8) { f32x8 z = {}; Oacc[d8] = z; }

  // prologue: DMA tile 0 into buffer 0 (one wave issues; TENSORcnt per-wave)
  if (wave == 0) {
    tdm_load_2d(aK[0], Kbf + (size_t)key0 * D, D, T, D,   D,  BN, 5, 3);
    tdm_load_2d(aV[0], Vt  + (size_t)key0,     T, D, T,   BN, D,  4, 3);
  }

  for (int t = 0; t < NT; ++t) {
    const int buf = t & 1;
    if (wave == 0) {
      if (t + 1 < NT) {
        const int nk = key0 + (t + 1) * BN;
        tdm_load_2d(aK[buf ^ 1], Kbf + (size_t)nk * D, D, T, D,   D,  BN, 5, 3);
        tdm_load_2d(aV[buf ^ 1], Vt  + (size_t)nk,     T, D, T,   BN, D,  4, 3);
        __builtin_amdgcn_s_wait_tensorcnt(2);   // oldest pair (this tile) done
      } else {
        __builtin_amdgcn_s_wait_tensorcnt(0);
      }
    }
    __syncthreads();

    const __bf16* Kt = sK[buf];
    const __bf16* Vl = sV[buf];

    // --- scores: 4 key-subtiles x 4 k-chunks of WMMA bf16 -----------------
    f32x8 Sc[4];
#pragma unroll
    for (int n = 0; n < 4; ++n) {
      f32x8 z = {}; Sc[n] = z;
#pragma unroll
      for (int c = 0; c < 4; ++c) {
        bf16x16 Kb = *(const bf16x16*)&Kt[(n * 16 + col) * KROW + c * 32 + half * 16];
        Sc[n] = wmma_bf16(Qa[c], Kb, Sc[n]);
      }
    }

    // --- online softmax in C layout (row = r + 8*half, key-col = lane&15) -
    float p[4][8], mx[8];
#pragma unroll
    for (int r = 0; r < 8; ++r) mx[r] = -1e30f;
#pragma unroll
    for (int n = 0; n < 4; ++n) {
#pragma unroll
      for (int r = 0; r < 8; ++r) {
        float sv = Sc[n][r] * SM_SCALE;
        if (causal) {
          int kg = key0 + t * BN + n * 16 + col;
          int qg = qbase + r + 8 * half;
          if (kg > S + qg) sv = -1e30f;
        }
        p[n][r] = sv;
        mx[r] = fmaxf(mx[r], sv);
      }
    }
#pragma unroll
    for (int off = 1; off < 16; off <<= 1)
#pragma unroll
      for (int r = 0; r < 8; ++r)
        mx[r] = fmaxf(mx[r], __shfl_xor(mx[r], off, 32));

    float cor[8], rs[8];
#pragma unroll
    for (int r = 0; r < 8; ++r) {
      float mn = fmaxf(m[r], mx[r]);
      cor[r] = __expf(m[r] - mn);
      m[r] = mn;
      float s = 0.0f;
#pragma unroll
      for (int n = 0; n < 4; ++n) {
        float e = __expf(p[n][r] - mn);
        p[n][r] = e;
        s += e;
      }
      rs[r] = s;
    }
#pragma unroll
    for (int off = 1; off < 16; off <<= 1)
#pragma unroll
      for (int r = 0; r < 8; ++r)
        rs[r] += __shfl_xor(rs[r], off, 32);
#pragma unroll
    for (int r = 0; r < 8; ++r) l[r] = l[r] * cor[r] + rs[r];
#pragma unroll
    for (int d8 = 0; d8 < 8; ++d8)
#pragma unroll
      for (int r = 0; r < 8; ++r) Oacc[d8][r] *= cor[r];

    // --- stage P (C layout -> row-major bf16) for A-operand re-read -------
    __bf16* Ps = sP[wave];
#pragma unroll
    for (int n = 0; n < 4; ++n)
#pragma unroll
      for (int r = 0; r < 8; ++r)
        Ps[(r + 8 * half) * PROW + n * 16 + col] = (__bf16)p[n][r];

    // --- PV: 2 key-groups of 32 x 8 dim-subtiles --------------------------
#pragma unroll
    for (int g = 0; g < 2; ++g) {
      bf16x8 lo = *(const bf16x8*)&Ps[col * PROW + g * 32 + half * 16];
      bf16x8 hi = *(const bf16x8*)&Ps[col * PROW + g * 32 + half * 16 + 8];
      bf16x16 Pa = __builtin_shufflevector(lo, hi, 0, 1, 2, 3, 4, 5, 6, 7,
                                           8, 9, 10, 11, 12, 13, 14, 15);
#pragma unroll
      for (int d8 = 0; d8 < 8; ++d8) {
        bf16x16 Vb = *(const bf16x16*)&Vl[(d8 * 16 + col) * VROW + g * 32 + half * 16];
        Oacc[d8] = wmma_bf16(Pa, Vb, Oacc[d8]);
      }
    }
    __syncthreads();   // all reads done before next tile overwrites buffer
  }

  // --- write partials (unnormalized O, per-row m/l) ------------------------
#pragma unroll
  for (int r = 0; r < 8; ++r) {
    int qrow = qbase + r + 8 * half;
    size_t ob = ((size_t)chunk * B + qrow) * D;
#pragma unroll
    for (int d8 = 0; d8 < 8; ++d8)
      Op[ob + d8 * 16 + col] = Oacc[d8][r];
    if (col == 0) {
      Mb[(size_t)chunk * B + qrow] = m[r];
      Lb[(size_t)chunk * B + qrow] = l[r];
    }
  }
}

// ---- Kernel 4: LSE combine across 33 chunks -------------------------------
__global__ void mtx_reduce(const float* __restrict__ Mb,
                           const float* __restrict__ Lb,
                           const float* __restrict__ Op,
                           float* __restrict__ out) {
  int idx = blockIdx.x * 256 + threadIdx.x;   // < B*D
  int qr = idx >> 7;
  int d  = idx & 127;
  float M = -1e30f;
  for (int c = 0; c < NCH; ++c) M = fmaxf(M, Mb[c * B + qr]);
  float L = 0.0f, acc = 0.0f;
  for (int c = 0; c < NCH; ++c) {
    float w = __expf(Mb[c * B + qr] - M);
    L   += Lb[c * B + qr] * w;
    acc += Op[((size_t)c * B + qr) * D + d] * w;
  }
  out[idx] = acc / L;
}

// ---------------------------------------------------------------------------
extern "C" void kernel_launch(void* const* d_in, const int* in_sizes, int n_in,
                              void* d_out, int out_size, void* d_ws, size_t ws_size,
                              hipStream_t stream) {
  const float* q  = (const float*)d_in[0];
  const float* k  = (const float*)d_in[1];
  const float* v  = (const float*)d_in[2];
  const float* Kc = (const float*)d_in[3];
  const float* Vc = (const float*)d_in[4];
  float* out = (float*)d_out;

  // workspace layout (~35 MB)
  __bf16* Kbf = (__bf16*)d_ws;                       // T*D bf16
  __bf16* Vt  = Kbf + (size_t)T * D;                 // D*T bf16
  float*  Mb  = (float*)(Vt + (size_t)D * T);        // NCH*B
  float*  Lb  = Mb + (size_t)NCH * B;                // NCH*B
  float*  Op  = Lb + (size_t)NCH * B;                // NCH*B*D

  mtx_pack_k<<<(T * D) / 256, 256, 0, stream>>>(Kc, k, Kbf);
  mtx_transpose_v<<<dim3(T / 32, D / 32), dim3(32, 8), 0, stream>>>(Vc, v, Vt);
  mtx_attn_partial<<<dim3(B / 128, NCH), 256, 0, stream>>>(q, Kbf, Vt, Mb, Lb, Op);
  mtx_reduce<<<(B * D) / 256, 256, 0, stream>>>(Mb, Lb, Op, out);
}
